// MultiHeadAttention_64682207477870
// MI455X (gfx1250) — matrix operand order repaired and checked by reading, not compiled, not run on hardware
//
#include <hip/hip_runtime.h>
#include <hip/hip_bf16.h>

// ---------------------------------------------------------------------------
// MHA for MI455X (gfx1250, wave32, WMMA f32<-f16 16x16x32)
// B=1, S=4096, E=768, H=12, D=64
// Pipeline: 3x GEMM (x@W^T+b -> f16 Q/K/V) ; fused flash-attention ;
//           GEMM (O@Wo^T+bo -> f32 out).  All WMMA, f32 accumulate.
// ---------------------------------------------------------------------------

typedef _Float16 half_t;
typedef __attribute__((ext_vector_type(16))) _Float16 v16h;
typedef __attribute__((ext_vector_type(8)))  _Float16 v8h;
typedef __attribute__((ext_vector_type(4)))  _Float16 v4h;
typedef __attribute__((ext_vector_type(8)))  float    v8f;
typedef __attribute__((ext_vector_type(4)))  int      v4i;

#define S_LEN 4096
#define E_DIM 768
#define NHEAD 12
#define HD    64

// ---- CDNA5 feature probes (compile-safe) ----------------------------------
#if __has_builtin(__builtin_amdgcn_global_load_async_to_lds_b128) && \
    __has_builtin(__builtin_amdgcn_s_wait_asynccnt)
#define HAVE_ASYNC_LDS 1
typedef __attribute__((address_space(1))) v4i gbl_v4i;
typedef __attribute__((address_space(3))) v4i lds_v4i;
#endif

__device__ __forceinline__ float fast_exp2(float x) {
#if __has_builtin(__builtin_amdgcn_exp2f)
  return __builtin_amdgcn_exp2f(x);
#else
  return exp2f(x);
#endif
}

// 16B global->LDS copy: async on CDNA5 toolchains, sync fallback otherwise.
__device__ __forceinline__ void copy16_g2l(const half_t* g, half_t* l) {
#if defined(HAVE_ASYNC_LDS)
  __builtin_amdgcn_global_load_async_to_lds_b128((gbl_v4i*)g, (lds_v4i*)l, 0, 0);
#else
  *(v8h*)l = *(const v8h*)g;
#endif
}

__device__ __forceinline__ void wait_async_copies() {
#if defined(HAVE_ASYNC_LDS)
  __builtin_amdgcn_s_wait_asynccnt(0);
#endif
}

// ---- WMMA wrapper ---------------------------------------------------------
__device__ __forceinline__ v8f wmma_f16(v16h a, v16h b, v8f c) {
  return __builtin_amdgcn_wmma_f32_16x16x32_f16(false, a, false, b, (short)0, c,
                                                false, false);
}

__device__ __forceinline__ v16h join8(v8h lo, v8h hi) {
  v16h r;
#pragma unroll
  for (int i = 0; i < 8; ++i) { r[i] = lo[i]; r[8 + i] = hi[i]; }
  return r;
}

// A fragment 16x32 (MxK) from row-major [m][k], stride ldm (ISA 7.12.2:
// lane&15 = M; lanes<16 hold K {0..7,16..23}, lanes>=16 K {8..15,24..31}).
__device__ __forceinline__ v16h load_frag_a(const half_t* p0, int ldm, int lane) {
  const half_t* p = p0 + (lane & 15) * ldm + ((lane >> 4) << 3);
  return join8(*(const v8h*)(p), *(const v8h*)(p + 16));
}

// B fragment 32x16 (KxN) from [n][k], stride ldn (lane&15 = N; lanes<16 hold
// K {0..7,16..23}, lanes>=16 hold K {8..15,24..31}).
__device__ __forceinline__ v16h load_frag_b(const half_t* p0, int ldn, int lane) {
  const half_t* p = p0 + (lane & 15) * ldn + ((lane >> 4) << 3);
  return join8(*(const v8h*)(p), *(const v8h*)(p + 16));
}

// ---------------------------------------------------------------------------
// GEMM: Y[M,N] = X[M,K] @ W[N,K]^T + bias, output scale.
// 256 threads (8 waves), 128x64 block tile, wave = 32x32, K-step 64
// -> 8 WMMAs per wave between each barrier pair.
// ---------------------------------------------------------------------------
#define LDG 72  // 64 + 8 halfs pad; 144B row stride (16B multiple)

template <typename TIn, typename TOut>
__global__ void __launch_bounds__(256)
gemm_xwt_bias(const TIn* __restrict__ X, const float* __restrict__ W,
              const float* __restrict__ bias, TOut* __restrict__ Y,
              int M, int N, int K, float outScale) {
  __shared__ half_t ldsA[128 * LDG];
  __shared__ half_t ldsB[64 * LDG];
  const int tid = threadIdx.x, lane = tid & 31, wave = tid >> 5;
  const int m0 = blockIdx.y * 128, n0 = blockIdx.x * 64;
  const int wm = (wave >> 1) * 32, wn = (wave & 1) * 32;
  const int sra = tid >> 1,  sca = (tid & 1) * 32;  // A: 128 rows x 64
  const int srb = tid >> 2,  scb = (tid & 3) * 16;  // B:  64 rows x 64
  v8f acc[2][2] = {};

  for (int k0 = 0; k0 < K; k0 += 64) {
    const TIn*   xa = X + (size_t)(m0 + sra) * K + k0 + sca;
    const float* wb = W + (size_t)(n0 + srb) * K + k0 + scb;
#pragma unroll
    for (int j = 0; j < 4; ++j) {
      v8h t;
#pragma unroll
      for (int i = 0; i < 8; ++i) t[i] = (half_t)(float)xa[j * 8 + i];
      *(v8h*)&ldsA[sra * LDG + sca + j * 8] = t;
    }
#pragma unroll
    for (int j = 0; j < 2; ++j) {
      v8h t;
#pragma unroll
      for (int i = 0; i < 8; ++i) t[i] = (half_t)wb[j * 8 + i];
      *(v8h*)&ldsB[srb * LDG + scb + j * 8] = t;
    }
    __syncthreads();
#pragma unroll
    for (int kd = 0; kd < 64; kd += 32) {
      v16h a0 = load_frag_a(&ldsA[(wm + 0)  * LDG + kd], LDG, lane);
      v16h a1 = load_frag_a(&ldsA[(wm + 16) * LDG + kd], LDG, lane);
      v16h b0 = load_frag_b(&ldsB[(wn + 0)  * LDG + kd], LDG, lane);
      v16h b1 = load_frag_b(&ldsB[(wn + 16) * LDG + kd], LDG, lane);
      acc[0][0] = wmma_f16(a0, b0, acc[0][0]);
      acc[0][1] = wmma_f16(a0, b1, acc[0][1]);
      acc[1][0] = wmma_f16(a1, b0, acc[1][0]);
      acc[1][1] = wmma_f16(a1, b1, acc[1][1]);
    }
    __syncthreads();
  }

  // C layout: lane&15 = N, reg r = M&7, lane>=16 -> M += 8.
#pragma unroll
  for (int mi = 0; mi < 2; ++mi)
#pragma unroll
    for (int ni = 0; ni < 2; ++ni) {
      const int nn  = n0 + wn + ni * 16 + (lane & 15);
      const float b = bias[nn];
      const int mr0 = m0 + wm + mi * 16 + ((lane >> 4) << 3);
#pragma unroll
      for (int r = 0; r < 8; ++r)
        Y[(size_t)(mr0 + r) * N + nn] = (TOut)((acc[mi][ni][r] + b) * outScale);
    }
}

// ---------------------------------------------------------------------------
// Flash attention, fused online softmax. Q carries log2(e)/sqrt(D) so all
// exponentials are raw exp2 (v_exp_f32). grid = (S/64, H); 128 threads
// (4 waves); wave owns 16 query rows; streams 64-key tiles through LDS.
// Per tile per wave: 8 WMMA (QK^T) + 8 WMMA (PV).
// ---------------------------------------------------------------------------
#define LDK 72   // K tile  [key][dim] stride
#define LDV 72   // Vt tile [dim][key] stride
#define LDP 72   // P scratch [m][key] stride

__global__ void __launch_bounds__(128)
attn_kernel(const half_t* __restrict__ Qh, const half_t* __restrict__ Kh,
            const half_t* __restrict__ Vh, half_t* __restrict__ Oh) {
  __shared__ half_t ldsK[64 * LDK];          // B for Q K^T
  __shared__ half_t ldsVt[HD * LDV];         // B for P V (transposed)
  __shared__ half_t ldsP[4 * 16 * LDP];      // per-wave P relayout scratch

  const int tid = threadIdx.x, lane = tid & 31, wave = tid >> 5;
  const int h = blockIdx.y;
  const int q0 = blockIdx.x * 64;
  const int hoff = h * HD;
  const int qrow = q0 + wave * 16;

  const half_t* qbase = Qh + (size_t)qrow * E_DIM + hoff;
  v16h aq0 = load_frag_a(qbase + 0,  E_DIM, lane);
  v16h aq1 = load_frag_a(qbase + 32, E_DIM, lane);

  v8f o[4] = {};
  float mrow[8], lrow[8];
#pragma unroll
  for (int r = 0; r < 8; ++r) { mrow[r] = -1e30f; lrow[r] = 0.0f; }

  half_t* myP = ldsP + wave * 16 * LDP;
  // K staging: thread t copies 32 halfs of one key row (async b128 x4).
  const int skey = tid >> 1, skc = (tid & 1) * 32;
  // V staging: thread t transposes a 4(key) x 8(dim) sub-tile.
  const int vkg = (tid >> 3) * 4, vdg = (tid & 7) * 8;

  for (int kt = 0; kt < S_LEN; kt += 64) {
    const half_t* ksrc = Kh + (size_t)(kt + skey) * E_DIM + hoff + skc;
#pragma unroll
    for (int j = 0; j < 4; ++j)
      copy16_g2l(ksrc + j * 8, &ldsK[skey * LDK + skc + j * 8]);

    v8h vr[4];
#pragma unroll
    for (int kk = 0; kk < 4; ++kk)
      vr[kk] = *(const v8h*)(Vh + (size_t)(kt + vkg + kk) * E_DIM + hoff + vdg);
#pragma unroll
    for (int i = 0; i < 8; ++i) {
      v4h p; p[0] = vr[0][i]; p[1] = vr[1][i]; p[2] = vr[2][i]; p[3] = vr[3][i];
      *(v4h*)&ldsVt[(vdg + i) * LDV + vkg] = p;
    }
    if (kt + 64 < S_LEN) {  // warm next tile (global_prefetch_b8)
      __builtin_prefetch(ksrc + (size_t)64 * E_DIM, 0, 1);
      __builtin_prefetch(Vh + (size_t)(kt + 64 + vkg) * E_DIM + hoff + vdg, 0, 1);
    }
    wait_async_copies();
    __syncthreads();

    // Scores: 16 rows x 64 keys = four 16x16 C fragments.
    v8f s[4];
#pragma unroll
    for (int nf = 0; nf < 4; ++nf) {
      v8f z = {};
      z = wmma_f16(aq0, load_frag_b(&ldsK[(nf * 16) * LDK + 0],  LDK, lane), z);
      z = wmma_f16(aq1, load_frag_b(&ldsK[(nf * 16) * LDK + 32], LDK, lane), z);
      s[nf] = z;
    }

    // Online softmax: row M lives across 16 lanes of one half-wave; xor
    // offsets 1,2,4,8 reduce within that half-wave only.
#pragma unroll
    for (int r = 0; r < 8; ++r) {
      float mx = fmaxf(fmaxf(s[0][r], s[1][r]), fmaxf(s[2][r], s[3][r]));
      mx = fmaxf(mx, __shfl_xor(mx, 1, 32));
      mx = fmaxf(mx, __shfl_xor(mx, 2, 32));
      mx = fmaxf(mx, __shfl_xor(mx, 4, 32));
      mx = fmaxf(mx, __shfl_xor(mx, 8, 32));
      const float mnew  = fmaxf(mrow[r], mx);
      const float alpha = fast_exp2(mrow[r] - mnew);
      float p0 = fast_exp2(s[0][r] - mnew);
      float p1 = fast_exp2(s[1][r] - mnew);
      float p2 = fast_exp2(s[2][r] - mnew);
      float p3 = fast_exp2(s[3][r] - mnew);
      float rs = (p0 + p1) + (p2 + p3);
      rs += __shfl_xor(rs, 1, 32);
      rs += __shfl_xor(rs, 2, 32);
      rs += __shfl_xor(rs, 4, 32);
      rs += __shfl_xor(rs, 8, 32);
      lrow[r] = lrow[r] * alpha + rs;
      mrow[r] = mnew;
      o[0][r] *= alpha; o[1][r] *= alpha; o[2][r] *= alpha; o[3][r] *= alpha;
      const int pm = ((lane >> 4) << 3) + r;   // C-frag -> [m][key] relayout
      myP[pm * LDP +  0 + (lane & 15)] = (half_t)p0;
      myP[pm * LDP + 16 + (lane & 15)] = (half_t)p1;
      myP[pm * LDP + 32 + (lane & 15)] = (half_t)p2;
      myP[pm * LDP + 48 + (lane & 15)] = (half_t)p3;
    }

    // P @ V: per-wave LDS is in-order (DS pipe) -> safe reload as A frags.
    v16h ap0 = load_frag_a(myP + 0,  LDP, lane);
    v16h ap1 = load_frag_a(myP + 32, LDP, lane);
#pragma unroll
    for (int f = 0; f < 4; ++f) {
      o[f] = wmma_f16(ap0, load_frag_b(&ldsVt[(f * 16) * LDV + 0],  LDV, lane), o[f]);
      o[f] = wmma_f16(ap1, load_frag_b(&ldsVt[(f * 16) * LDV + 32], LDV, lane), o[f]);
    }
    __syncthreads();  // protect ldsK/ldsVt before next tile's staging
  }

  // Normalize, store O in [s, E] (heads re-interleaved), f16.
  float invl[8];
#pragma unroll
  for (int r = 0; r < 8; ++r) invl[r] = 1.0f / lrow[r];
#pragma unroll
  for (int f = 0; f < 4; ++f) {
    const int col  = hoff + f * 16 + (lane & 15);
    const int row0 = qrow + ((lane >> 4) << 3);
#pragma unroll
    for (int r = 0; r < 8; ++r)
      Oh[(size_t)(row0 + r) * E_DIM + col] = (half_t)(o[f][r] * invl[r]);
  }
}

// ---------------------------------------------------------------------------
// Host launcher
// ---------------------------------------------------------------------------
extern "C" void kernel_launch(void* const* d_in, const int* in_sizes, int n_in,
                              void* d_out, int out_size, void* d_ws, size_t ws_size,
                              hipStream_t stream) {
  (void)in_sizes; (void)n_in; (void)out_size; (void)ws_size;

  const float* x  = (const float*)d_in[0];
  const float* Wq = (const float*)d_in[1];
  const float* bq = (const float*)d_in[2];
  const float* Wk = (const float*)d_in[3];
  const float* bk = (const float*)d_in[4];
  const float* Wv = (const float*)d_in[5];
  const float* bv = (const float*)d_in[6];
  const float* Wo = (const float*)d_in[7];
  const float* bo = (const float*)d_in[8];
  float* out = (float*)d_out;

  const size_t SE = (size_t)S_LEN * E_DIM;
  half_t* Qh = (half_t*)d_ws;      // f16 workspace: Q,K,V,O -> ~25.2 MB
  half_t* Kh = Qh + SE;
  half_t* Vh = Kh + SE;
  half_t* Oh = Vh + SE;

  const dim3 gProj(E_DIM / 64, S_LEN / 128);  // 12 x 32
  // 1/sqrt(D) * log2(e): scores land in log2 domain -> softmax uses raw exp2.
  const float qScale = 0.125f * 1.44269504088896340736f;

  gemm_xwt_bias<float, half_t><<<gProj, 256, 0, stream>>>(
      x, Wq, bq, Qh, S_LEN, E_DIM, E_DIM, qScale);
  gemm_xwt_bias<float, half_t><<<gProj, 256, 0, stream>>>(
      x, Wk, bk, Kh, S_LEN, E_DIM, E_DIM, 1.0f);
  gemm_xwt_bias<float, half_t><<<gProj, 256, 0, stream>>>(
      x, Wv, bv, Vh, S_LEN, E_DIM, E_DIM, 1.0f);

  const dim3 gAttn(S_LEN / 64, NHEAD);        // 64 x 12
  attn_kernel<<<gAttn, 128, 0, stream>>>(Qh, Kh, Vh, Oh);

  gemm_xwt_bias<half_t, float><<<gProj, 256, 0, stream>>>(
      Oh, Wo, bo, out, S_LEN, E_DIM, E_DIM, 1.0f);
}
